// CosineAttention_1967095021749
// MI455X (gfx1250) — matrix operand order repaired and checked by reading, not compiled, not run on hardware
//
#include <hip/hip_runtime.h>
#include <hip/hip_bf16.h>
#include <math.h>

typedef __attribute__((ext_vector_type(16))) _Float16 v16h;
typedef __attribute__((ext_vector_type(8)))  _Float16 v8h;
typedef __attribute__((ext_vector_type(8)))  float    v8f;
typedef __attribute__((ext_vector_type(4)))  float    v4f;

__device__ __forceinline__ v8f wmma16(v16h a, v16h b, v8f c) {
  // v_wmma_f32_16x16x32_f16
  return __builtin_amdgcn_wmma_f32_16x16x32_f16(false, a, false, b, (short)0, c, false, false);
}

// A fragment (16x32 f16). ISA layout: lane holds row m = lane&15; element j -> K = (lane>>4)*8 + (j&7) + (j>>3)*16.
// Caller passes p = &row[(lane>>4)*8]; halves at p[0..7] and p[16..23]. Two b128 loads.
__device__ __forceinline__ v16h load_frag_a(const _Float16* p) {
  v8h lo = *(const v8h*)p;
  v8h hv = *(const v8h*)(p + 16);
  v16h r;
#pragma unroll
  for (int j = 0; j < 8; ++j) { r[j] = lo[j]; r[j + 8] = hv[j]; }
  return r;
}

// B fragment (32x16 f16). Lane holds col n = lane&15; element j -> K = (lane>>4)*8 + (j&7) + (j>>3)*16.
// Caller stores operand as [n][k] (k contiguous) and passes p = &Bs[n][(lane>>4)*8]. Two b128 loads.
__device__ __forceinline__ v16h load_frag_b(const _Float16* p) {
  v8h lo = *(const v8h*)p;
  v8h hv = *(const v8h*)(p + 16);
  v16h r;
#pragma unroll
  for (int j = 0; j < 8; ++j) { r[j] = lo[j]; r[j + 8] = hv[j]; }
  return r;
}

// CDNA5 async global->LDS copy: 32 bytes per lane (two B128 transfers, ASYNCcnt).
// INST_OFFSET is added to BOTH the LDS and global addresses (ISA 10.x async pseudocode),
// so the second transfer reuses the same address VGPRs with offset:16.
// LDS byte address = low 32 bits of the flat shared pointer (aperture rule).
__device__ __forceinline__ void async_copy32(const _Float16* gsrc, _Float16* lds_dst) {
  unsigned l = (unsigned)(uintptr_t)lds_dst;
  asm volatile("global_load_async_to_lds_b128 %0, %1, off"
               :: "v"(l), "v"(gsrc) : "memory");
  asm volatile("global_load_async_to_lds_b128 %0, %1, off offset:16"
               :: "v"(l), "v"(gsrc) : "memory");
}

__device__ __forceinline__ void wait_async0() {
  asm volatile("s_wait_asynccnt 0x0" ::: "memory");
}

// ---------------------------------------------------------------------------
// Kernel A: magnitude-preserving weight normalization -> f16.
// wn[o][i] = w[o][i] / (EPS*sqrt(I) + ||w[o]||).  One wave per output row.
// ---------------------------------------------------------------------------
__global__ __launch_bounds__(128) void wnorm_kernel(const float* __restrict__ W,
                                                    _Float16* __restrict__ Wn, int I) {
  int row  = blockIdx.x * 4 + (threadIdx.x >> 5);
  int lane = threadIdx.x & 31;
  const float* wr = W + (size_t)row * I;
  float ss = 0.f;
  for (int i = lane; i < I; i += 32) { float wv = wr[i]; ss += wv * wv; }
  ss += __shfl_xor(ss, 1);  ss += __shfl_xor(ss, 2);
  ss += __shfl_xor(ss, 4);  ss += __shfl_xor(ss, 8);  ss += __shfl_xor(ss, 16);
  float inv = 1.f / (1e-4f * sqrtf((float)I) + sqrtf(ss));
  for (int i = lane; i < I; i += 32)
    Wn[(size_t)row * I + i] = (_Float16)(wr[i] * inv);
}

// ---------------------------------------------------------------------------
// Kernel B: QKV projection GEMM.  Y[b] = Wn[1536,512] @ X[b][512,1024], f16 out.
// WG = 256 threads (8 waves, 2x4), WG tile 128x256, wave tile 64x64, K-step 32.
// A tile staged via async global->LDS; B tile converted f32->f16 + transposed by VALU
// (overlaps with the in-flight async copy).
// ---------------------------------------------------------------------------
__global__ __launch_bounds__(256) void qkv_gemm_kernel(
    const _Float16* __restrict__ A,   // [1536][512] normalized f16
    const float*    __restrict__ X,   // [16][512][1024] f32
    _Float16*       __restrict__ Y) { // [16][1536][1024] f16
  const int M = 1536, K = 512, N = 1024;
  __shared__ __align__(16) _Float16 As[128][32];
  __shared__ __align__(16) _Float16 Bs[256][32];   // [n][k] transposed
  const int tilesN = N / 256, tilesM = M / 128;
  int bid = blockIdx.x;
  int b   = bid / (tilesM * tilesN);
  int rem = bid % (tilesM * tilesN);
  int m0  = (rem / tilesN) * 128;
  int n0  = (rem % tilesN) * 256;
  int tid = threadIdx.x, lane = tid & 31, wid = tid >> 5;
  int l15 = lane & 15, hi = lane >> 4;
  int wm = (wid & 1) * 64, wn = (wid >> 1) * 64;
  const float* Xb = X + (size_t)b * K * N;
  v8f zero8 = {0.f,0.f,0.f,0.f,0.f,0.f,0.f,0.f};
  v8f acc[4][4];
#pragma unroll
  for (int i = 0; i < 4; ++i)
#pragma unroll
    for (int j = 0; j < 4; ++j) acc[i][j] = zero8;

  int arow = tid >> 1, ach = (tid & 1) * 16;   // A-tile copy assignment
  for (int k0 = 0; k0 < K; k0 += 32) {
    __syncthreads();
    // A tile: async global->LDS (pure copy, 32B/lane), overlapped with B transform
    async_copy32(A + (size_t)(m0 + arow) * K + k0 + ach, &As[arow][ach]);
    // B tile: f32 load (b128), convert, transposed f16 store to [n][k]
#pragma unroll
    for (int t = 0; t < 8; ++t) {
      int idx = tid * 8 + t;          // 0..2047
      int kr  = idx >> 6;             // 0..31
      int nc  = (idx & 63) * 4;       // 0..252
      v4f xv = *(const v4f*)(Xb + (size_t)(k0 + kr) * N + n0 + nc);
#pragma unroll
      for (int j = 0; j < 4; ++j) Bs[nc + j][kr] = (_Float16)xv[j];
    }
    wait_async0();
    __syncthreads();
    v16h af[4], bf[4];
#pragma unroll
    for (int mi = 0; mi < 4; ++mi) af[mi] = load_frag_a(&As[wm + mi * 16 + l15][hi * 8]);
#pragma unroll
    for (int ni = 0; ni < 4; ++ni) bf[ni] = load_frag_b(&Bs[wn + ni * 16 + l15][hi * 8]);
#pragma unroll
    for (int mi = 0; mi < 4; ++mi)
#pragma unroll
      for (int ni = 0; ni < 4; ++ni)
        acc[mi][ni] = wmma16(af[mi], bf[ni], acc[mi][ni]);
  }
  _Float16* Yb = Y + (size_t)b * M * N;
#pragma unroll
  for (int mi = 0; mi < 4; ++mi)
#pragma unroll
    for (int ni = 0; ni < 4; ++ni)
#pragma unroll
      for (int r = 0; r < 8; ++r) {
        int row = m0 + wm + mi * 16 + r + hi * 8;   // D layout: M = r + 8*hi
        int col = n0 + wn + ni * 16 + l15;
        Yb[(size_t)row * N + col] = (_Float16)acc[mi][ni][r];
      }
}

// ---------------------------------------------------------------------------
// Kernel C: pixel-norm across the 8 heads for q/k/v + layout split.
// Q,K -> [bh][s][64] f16 (1/sqrt(64) folded into Q); V -> [bh][64][s] f16 (transposed).
// ---------------------------------------------------------------------------
__global__ void norm_split_kernel(const _Float16* __restrict__ Y,
                                  _Float16* __restrict__ Qh,
                                  _Float16* __restrict__ Kh,
                                  _Float16* __restrict__ Vt) {
  int bs = blockIdx.x;            // b*1024 + s
  int b = bs >> 10, s = bs & 1023;
  int d = threadIdx.x;            // 0..63
  const _Float16* Yb = Y + (size_t)b * 1536 * 1024;
  float q[8], k[8], v[8];
  float sq = 0.f, sk = 0.f, sv = 0.f;
#pragma unroll
  for (int h = 0; h < 8; ++h) {
    q[h] = (float)Yb[(size_t)(h * 64 + d) * 1024 + s];          sq += q[h] * q[h];
    k[h] = (float)Yb[(size_t)(512 + h * 64 + d) * 1024 + s];    sk += k[h] * k[h];
    v[h] = (float)Yb[(size_t)(1024 + h * 64 + d) * 1024 + s];   sv += v[h] * v[h];
  }
  float rq = rsqrtf(sq * 0.125f + 1e-4f) * 0.125f;  // pixel_norm * 1/sqrt(head_dim)
  float rk = rsqrtf(sk * 0.125f + 1e-4f);
  float rv = rsqrtf(sv * 0.125f + 1e-4f);
#pragma unroll
  for (int h = 0; h < 8; ++h) {
    size_t bh = (size_t)(b * 8 + h);
    Qh[bh * 65536 + (size_t)s * 64 + d] = (_Float16)(q[h] * rq);
    Kh[bh * 65536 + (size_t)s * 64 + d] = (_Float16)(k[h] * rk);
    Vt[bh * 65536 + (size_t)d * 1024 + s] = (_Float16)(v[h] * rv);
  }
}

// ---------------------------------------------------------------------------
// Kernel D: flash-style attention per (b,head). 8 waves x 16 q-rows = 128 q/WG,
// key blocks of 64. K/V tiles staged via async global->LDS. Online softmax across
// the 16-lane halves (matches C/D layout). Output written transposed as
// res[b][h*64+d][s] f16.
// ---------------------------------------------------------------------------
__global__ __launch_bounds__(256) void attn_kernel(
    const _Float16* __restrict__ Qh, const _Float16* __restrict__ Kh,
    const _Float16* __restrict__ Vt, _Float16* __restrict__ Res) {
  const int SEQ = 1024, HD = 64, BK = 64;
  __shared__ __align__(16) _Float16 Ks[64][64];      // [key][d]
  __shared__ __align__(16) _Float16 Vs[64][64];      // [d][key]
  __shared__ __align__(16) _Float16 Ps[8][16][64];   // per-wave P tile [m][key]
  int bh = blockIdx.x >> 3;
  int qb = blockIdx.x & 7;
  int b = bh >> 3, h = bh & 7;
  int tid = threadIdx.x, wid = tid >> 5, lane = tid & 31;
  int l15 = lane & 15, hi = lane >> 4;
  const _Float16* Qb = Qh + (size_t)bh * SEQ * HD;
  const _Float16* Kb = Kh + (size_t)bh * SEQ * HD;
  const _Float16* Vb = Vt + (size_t)bh * HD * SEQ;
  int qrow = qb * 128 + wid * 16 + l15;
  v16h qf[2];
  qf[0] = load_frag_a(Qb + (size_t)qrow * HD +      hi * 8);
  qf[1] = load_frag_a(Qb + (size_t)qrow * HD + 32 + hi * 8);
  v8f zero8 = {0.f,0.f,0.f,0.f,0.f,0.f,0.f,0.f};
  v8f o[4];
  float m_run[8], l_run[8];
#pragma unroll
  for (int i = 0; i < 4; ++i) o[i] = zero8;
#pragma unroll
  for (int r = 0; r < 8; ++r) { m_run[r] = -3.0e38f; l_run[r] = 0.f; }

  int crow = tid >> 2, cch = (tid & 3) * 16;   // tile copy assignment
  for (int kb = 0; kb < SEQ; kb += BK) {
    __syncthreads();
    // K (row-major [key][d]) and V^T (row-major [d][key]) tiles: async copies
    async_copy32(Kb + (size_t)(kb + crow) * HD + cch, &Ks[crow][cch]);
    async_copy32(Vb + (size_t)crow * SEQ + kb + cch,  &Vs[crow][cch]);
    wait_async0();
    __syncthreads();
    // S = Q @ K^T : 16 x 64 per wave
    v8f s[4];
#pragma unroll
    for (int ni = 0; ni < 4; ++ni) {
      s[ni] = zero8;
#pragma unroll
      for (int kk = 0; kk < 2; ++kk) {
        v16h bf = load_frag_b(&Ks[ni * 16 + l15][kk * 32 + hi * 8]);
        s[ni] = wmma16(qf[kk], bf, s[ni]);
      }
    }
    // Online softmax per row (row m = r + 8*hi, spread over 16-lane half)
#pragma unroll
    for (int r = 0; r < 8; ++r) {
      float mx = s[0][r];
#pragma unroll
      for (int ni = 1; ni < 4; ++ni) mx = fmaxf(mx, s[ni][r]);
      mx = fmaxf(mx, __shfl_xor(mx, 1)); mx = fmaxf(mx, __shfl_xor(mx, 2));
      mx = fmaxf(mx, __shfl_xor(mx, 4)); mx = fmaxf(mx, __shfl_xor(mx, 8));
      float nm = fmaxf(m_run[r], mx);
      float sc = __expf(m_run[r] - nm);
      m_run[r] = nm;
      float rsum = 0.f;
#pragma unroll
      for (int ni = 0; ni < 4; ++ni) {
        float p = __expf(s[ni][r] - nm);
        rsum += p;
        Ps[wid][r + hi * 8][ni * 16 + l15] = (_Float16)p;
      }
      rsum += __shfl_xor(rsum, 1); rsum += __shfl_xor(rsum, 2);
      rsum += __shfl_xor(rsum, 4); rsum += __shfl_xor(rsum, 8);
      l_run[r] = l_run[r] * sc + rsum;
#pragma unroll
      for (int ni = 0; ni < 4; ++ni) o[ni][r] *= sc;
    }
    // Per-wave LDS ordering guarantees store->load visibility; wait DS explicitly.
    asm volatile("s_wait_dscnt 0" ::: "memory");
    // O += P @ V : A = P (16 x 64 keys) from LDS, B = V^T rows (contiguous keys)
#pragma unroll
    for (int kk = 0; kk < 2; ++kk) {
      v16h pf = load_frag_a(&Ps[wid][l15][kk * 32 + hi * 8]);
#pragma unroll
      for (int ni = 0; ni < 4; ++ni) {
        v16h vf = load_frag_b(&Vs[ni * 16 + l15][kk * 32 + hi * 8]);
        o[ni] = wmma16(pf, vf, o[ni]);
      }
    }
  }
  // Normalize and emit transposed: res[b][h*64+d][s]
#pragma unroll
  for (int r = 0; r < 8; ++r) {
    float inv = 1.f / l_run[r];
    int srow = qb * 128 + wid * 16 + r + hi * 8;
#pragma unroll
    for (int ni = 0; ni < 4; ++ni) {
      int d = ni * 16 + l15;
      Res[((size_t)b * 512 + h * 64 + d) * 1024 + srow] = (_Float16)(o[ni][r] * inv);
    }
  }
}

// ---------------------------------------------------------------------------
// Kernel E: projection GEMM + fused mp_add epilogue.
// out = ((1-t)*x + t*(Wp @ res)) / sqrt(t^2+(1-t)^2), f32 out.
// ---------------------------------------------------------------------------
__global__ __launch_bounds__(256) void proj_gemm_kernel(
    const _Float16* __restrict__ A,   // [512][512] normalized f16
    const _Float16* __restrict__ R,   // [16][512][1024] f16 (res)
    const float*    __restrict__ X,   // [16][512][1024] f32 (skip input)
    float*          __restrict__ Out) {
  const int M = 512, K = 512, N = 1024;
  __shared__ __align__(16) _Float16 As[128][32];
  __shared__ __align__(16) _Float16 Bs[256][32];
  int bid = blockIdx.x;
  int b   = bid >> 4;
  int rem = bid & 15;
  int m0  = (rem >> 2) * 128;
  int n0  = (rem & 3) * 256;
  int tid = threadIdx.x, lane = tid & 31, wid = tid >> 5;
  int l15 = lane & 15, hi = lane >> 4;
  int wm = (wid & 1) * 64, wn = (wid >> 1) * 64;
  const _Float16* Rb = R + (size_t)b * K * N;
  v8f zero8 = {0.f,0.f,0.f,0.f,0.f,0.f,0.f,0.f};
  v8f acc[4][4];
#pragma unroll
  for (int i = 0; i < 4; ++i)
#pragma unroll
    for (int j = 0; j < 4; ++j) acc[i][j] = zero8;

  int arow = tid >> 1, ach = (tid & 1) * 16;
  for (int k0 = 0; k0 < K; k0 += 32) {
    __syncthreads();
    // A tile: async global->LDS copy
    async_copy32(A + (size_t)(m0 + arow) * K + k0 + ach, &As[arow][ach]);
    // B tile: f16 transpose into [n][k]
#pragma unroll
    for (int t = 0; t < 4; ++t) {
      int idx = tid * 4 + t;          // 0..1023
      int kr  = idx >> 5;             // 0..31
      int nc  = (idx & 31) * 8;       // 0..248
      v8h rv = *(const v8h*)(Rb + (size_t)(k0 + kr) * N + n0 + nc);
#pragma unroll
      for (int j = 0; j < 8; ++j) Bs[nc + j][kr] = rv[j];
    }
    wait_async0();
    __syncthreads();
    v16h af[4], bf[4];
#pragma unroll
    for (int mi = 0; mi < 4; ++mi) af[mi] = load_frag_a(&As[wm + mi * 16 + l15][hi * 8]);
#pragma unroll
    for (int ni = 0; ni < 4; ++ni) bf[ni] = load_frag_b(&Bs[wn + ni * 16 + l15][hi * 8]);
#pragma unroll
    for (int mi = 0; mi < 4; ++mi)
#pragma unroll
      for (int ni = 0; ni < 4; ++ni)
        acc[mi][ni] = wmma16(af[mi], bf[ni], acc[mi][ni]);
  }
  // mp_add epilogue: t=0.3, scale = sqrt(0.58)
  const float A0 = 0.91914503f;   // 0.7 / sqrt(0.58)
  const float A1 = 0.39391930f;   // 0.3 / sqrt(0.58)
  const float* Xb = X + (size_t)b * M * N;
  float* Ob = Out + (size_t)b * M * N;
#pragma unroll
  for (int mi = 0; mi < 4; ++mi)
#pragma unroll
    for (int ni = 0; ni < 4; ++ni)
#pragma unroll
      for (int r = 0; r < 8; ++r) {
        int row = m0 + wm + mi * 16 + r + hi * 8;
        int col = n0 + wn + ni * 16 + l15;
        size_t off = (size_t)row * N + col;
        Ob[off] = A0 * Xb[off] + A1 * acc[mi][ni][r];
      }
}

// ---------------------------------------------------------------------------
extern "C" void kernel_launch(void* const* d_in, const int* in_sizes, int n_in,
                              void* d_out, int out_size, void* d_ws, size_t ws_size,
                              hipStream_t stream) {
  (void)in_sizes; (void)n_in; (void)out_size; (void)ws_size;
  const float* x      = (const float*)d_in[0];   // [16][512][32][32]
  const float* w_attn = (const float*)d_in[1];   // [1536][512][1][1]
  const float* w_proj = (const float*)d_in[2];   // [512][512][1][1]
  float* out = (float*)d_out;

  // Workspace layout (f16 elements)
  _Float16* ws  = (_Float16*)d_ws;
  _Float16* wA  = ws;                       // 1536*512      =   786,432
  _Float16* wP  = wA + 786432;              // 512*512       =   262,144
  _Float16* Y   = wP + 262144;              // 16*1536*1024  = 25,165,824
  _Float16* Qh  = Y  + 25165824;            // 16*8*1024*64  =  8,388,608
  _Float16* Kh  = Qh + 8388608;
  _Float16* Vt  = Kh + 8388608;
  _Float16* Res = Vt + 8388608;             // 16*512*1024   =  8,388,608

  wnorm_kernel<<<1536 / 4, 128, 0, stream>>>(w_attn, wA, 512);
  wnorm_kernel<<<512 / 4, 128, 0, stream>>>(w_proj, wP, 512);
  qkv_gemm_kernel<<<16 * 12 * 4, 256, 0, stream>>>(wA, x, Y);
  norm_split_kernel<<<16 * 1024, 64, 0, stream>>>(Y, Qh, Kh, Vt);
  attn_kernel<<<128 * 8, 256, 0, stream>>>(Qh, Kh, Vt, Res);
  proj_gemm_kernel<<<16 * 16, 256, 0, stream>>>(wP, Res, x, out);
}
